// Encoder_15135464751432
// MI455X (gfx1250) — matrix operand order repaired and checked by reading, not compiled, not run on hardware
//
#include <hip/hip_runtime.h>

typedef __attribute__((ext_vector_type(16))) _Float16 v16h;
typedef __attribute__((ext_vector_type(8)))  float    v8f;

#define NEG_SLOPE 0.1f

// ---------------- degree / norm ----------------
__global__ void k_init_deg(float* deg, int N) {
  int i = blockIdx.x * blockDim.x + threadIdx.x;
  if (i < N) deg[i] = 1.0f;  // self-loop contributes 1
}

__global__ void k_count_deg(const long long* dst, float* deg, int E) {
  int e = blockIdx.x * blockDim.x + threadIdx.x;
  if (e < E) unsafeAtomicAdd(&deg[(int)dst[e]], 1.0f);  // global_atomic_add_f32
}

__global__ void k_rsqrt(float* deg, int N) {
  int i = blockIdx.x * blockDim.x + threadIdx.x;
  if (i < N) deg[i] = rsqrtf(deg[i]);  // in-place: deg -> dinv
}

// ---------------- propagation hop ----------------
// h_new[i] = dinv[i]^2 * h_old[i]   (self-loop term; fully initializes h_new)
__global__ void k_selfloop(const float* __restrict__ hold, const float* __restrict__ dinv,
                           float* __restrict__ hnew, int N) {
  int t  = blockIdx.x * blockDim.x + threadIdx.x;  // N*32 threads, float4 per thread
  int i  = t >> 5;
  int c4 = (t & 31) << 2;
  if (i < N) {
    float w = dinv[i] * dinv[i];
    const float4 v = *(const float4*)(hold + (size_t)i * 128 + c4);
    float4 o; o.x = w * v.x; o.y = w * v.y; o.z = w * v.z; o.w = w * v.w;
    *(float4*)(hnew + (size_t)i * 128 + c4) = o;
  }
}

// h_new[dst] += dinv[src]*dinv[dst] * h_old[src]; 32 lanes per edge, float4 per lane
__global__ void k_edge_scatter(const long long* __restrict__ ei, const float* __restrict__ dinv,
                               const float* __restrict__ hold, float* __restrict__ hnew, int E) {
  int t  = blockIdx.x * blockDim.x + threadIdx.x;
  int e  = t >> 5;
  int c4 = (t & 31) << 2;
  if (e < E) {
    int s = (int)ei[e];
    int d = (int)ei[(size_t)E + e];
    float nrm = dinv[s] * dinv[d];
    const float4 v = *(const float4*)(hold + (size_t)s * 128 + c4);  // 512B/wave coalesced gather
    float* o = hnew + (size_t)d * 128 + c4;
    unsafeAtomicAdd(o + 0, nrm * v.x);
    unsafeAtomicAdd(o + 1, nrm * v.y);
    unsafeAtomicAdd(o + 2, nrm * v.z);
    unsafeAtomicAdd(o + 3, nrm * v.w);
  }
}

// ---------------- WMMA packing ----------------
// B (=W^T) tile layout for v_wmma_f32_16x16x32_f16, wave32:
//   lane l: col N = l%16 ; halves e: K = (e/8)*16 + (l/16)*8 + e%8
//   packed as [(nt*ksteps + ks)*32 + lane][16 halves]  (32B contiguous per lane)
__global__ void k_pack_W(const float* __restrict__ W, _Float16* __restrict__ Wp,
                         int Cout, int Cin) {
  int t = blockIdx.x * blockDim.x + threadIdx.x;
  int ksteps = Cin >> 5;
  int total = (Cout >> 4) * ksteps * 512;
  if (t >= total) return;
  int e    = t & 15;
  int lane = (t >> 4) & 31;
  int rest = t >> 9;
  int ks   = rest % ksteps;
  int nt   = rest / ksteps;
  int n = nt * 16 + (lane & 15);
  int k = ks * 32 + ((e >> 3) << 4) + ((lane >> 4) << 3) + (e & 7);
  Wp[t] = (_Float16)W[(size_t)n * Cin + k];  // B[k][n] = W[n][k]
}

// A tile layout: lane l: row M = l%16 ; halves e: K = ks*32 + (e/8)*16 + (l/16)*8 + e%8
// mtilesPad >= mtiles (even): pad tiles are zero-filled so 2-wide M blocking is safe.
__global__ void k_pack_A(const float* __restrict__ h, _Float16* __restrict__ Ap,
                         int N, int Cin, int mtilesPad) {
  int t = blockIdx.x * blockDim.x + threadIdx.x;
  int ksteps = Cin >> 5;
  int total = mtilesPad * ksteps * 512;
  if (t >= total) return;
  int e    = t & 15;
  int lane = (t >> 4) & 31;
  int rest = t >> 9;
  int ks   = rest % ksteps;
  int mt   = rest / ksteps;
  int M  = mt * 16 + (lane & 15);
  int hi = lane >> 4;
  int k  = ks * 32 + ((e >> 3) << 4) + (hi << 3) + (e & 7);
  Ap[t] = (M < N) ? (_Float16)h[(size_t)M * Cin + k] : (_Float16)0.0f;
}

// ---------------- GEMM: one wave per 32x64 output block (2x4 WMMA tiles) ----------------
// Per K-step: 2 A-frags + 4 B-frags (12 b128 loads) feed 8 WMMAs; A reused 4x, B 2x.
__global__ void k_gemm_wmma(const _Float16* __restrict__ Ap, const _Float16* __restrict__ Wp,
                            const float* __restrict__ bias, float* __restrict__ out,
                            int N, int Cout, int Cin, int mpairs) {
  int wave = (blockIdx.x * blockDim.x + threadIdx.x) >> 5;
  int lane = threadIdx.x & 31;
  int nquads = Cout >> 6;          // 64-wide N strips
  int mp = wave / nquads;
  int nq = wave % nquads;
  if (mp >= mpairs) return;        // wave-uniform: EXEC stays all-1s for WMMA
  int ksteps = Cin >> 5;
  int mt0 = mp << 1;
  int nt0 = nq << 2;

  v8f acc[2][4] = {};
  const v16h* A = (const v16h*)Ap;
  const v16h* B = (const v16h*)Wp;

  for (int ks = 0; ks < ksteps; ++ks) {
    v16h a0 = A[(size_t)((mt0 + 0) * ksteps + ks) * 32 + lane];
    v16h a1 = A[(size_t)((mt0 + 1) * ksteps + ks) * 32 + lane];
    v16h b0 = B[(size_t)((nt0 + 0) * ksteps + ks) * 32 + lane];
    v16h b1 = B[(size_t)((nt0 + 1) * ksteps + ks) * 32 + lane];
    v16h b2 = B[(size_t)((nt0 + 2) * ksteps + ks) * 32 + lane];
    v16h b3 = B[(size_t)((nt0 + 3) * ksteps + ks) * 32 + lane];
    acc[0][0] = __builtin_amdgcn_wmma_f32_16x16x32_f16(false, a0, false, b0, (short)0, acc[0][0], false, false);
    acc[0][1] = __builtin_amdgcn_wmma_f32_16x16x32_f16(false, a0, false, b1, (short)0, acc[0][1], false, false);
    acc[0][2] = __builtin_amdgcn_wmma_f32_16x16x32_f16(false, a0, false, b2, (short)0, acc[0][2], false, false);
    acc[0][3] = __builtin_amdgcn_wmma_f32_16x16x32_f16(false, a0, false, b3, (short)0, acc[0][3], false, false);
    acc[1][0] = __builtin_amdgcn_wmma_f32_16x16x32_f16(false, a1, false, b0, (short)0, acc[1][0], false, false);
    acc[1][1] = __builtin_amdgcn_wmma_f32_16x16x32_f16(false, a1, false, b1, (short)0, acc[1][1], false, false);
    acc[1][2] = __builtin_amdgcn_wmma_f32_16x16x32_f16(false, a1, false, b2, (short)0, acc[1][2], false, false);
    acc[1][3] = __builtin_amdgcn_wmma_f32_16x16x32_f16(false, a1, false, b3, (short)0, acc[1][3], false, false);
  }

  // C/D layout: lane -> col = lane%16 ; VGPR r -> row = 16*tile + 8*(lane/16) + r
  #pragma unroll
  for (int tm = 0; tm < 2; ++tm) {
    int rbase = (mt0 + tm) * 16 + ((lane >> 4) << 3);
    #pragma unroll
    for (int tn = 0; tn < 4; ++tn) {
      int col = (nt0 + tn) * 16 + (lane & 15);
      float bb = bias[col];
      #pragma unroll
      for (int r = 0; r < 8; ++r) {
        int row = rbase + r;
        if (row < N) {
          float y = acc[tm][tn][r] + bb;
          out[(size_t)row * Cout + col] = (y >= 0.0f) ? y : NEG_SLOPE * y;
        }
      }
    }
  }
}

extern "C" void kernel_launch(void* const* d_in, const int* in_sizes, int n_in,
                              void* d_out, int out_size, void* d_ws, size_t ws_size,
                              hipStream_t stream) {
  const float*     x    = (const float*)d_in[0];
  const long long* ei   = (const long long*)d_in[1];
  const float*     W    = (const float*)d_in[2];
  const float*     bias = (const float*)d_in[3];
  float*           out  = (float*)d_out;

  const int Cout = in_sizes[3];              // 512
  const int Cin  = in_sizes[2] / Cout;       // 128
  const int N    = in_sizes[0] / Cin;        // 50000
  const int E    = in_sizes[1] / 2;          // 800000
  const int mtiles    = (N + 15) >> 4;
  const int mtilesPad = (mtiles + 1) & ~1;   // even for 2-wide M blocking
  const int ntiles    = Cout >> 4;
  const int ksteps    = Cin >> 5;

  // workspace layout: [deg/dinv | Wpack | Apack]
  char* ws = (char*)d_ws;
  float*     deg = (float*)ws;
  size_t off = ((size_t)N * 4 + 255) & ~(size_t)255;
  _Float16*  Wp  = (_Float16*)(ws + off);
  off += ((size_t)Cout * Cin * 2 + 255) & ~(size_t)255;
  _Float16*  Ap  = (_Float16*)(ws + off);

  // h ping-pong buffers inside d_out (out is N*Cout floats >= 2*N*Cin)
  float* h1 = out;
  float* h2 = out + (size_t)N * Cin;

  const int T = 256;
  // 1) degrees + rsqrt (in-place -> dinv)
  k_init_deg <<<(N + T - 1) / T, T, 0, stream>>>(deg, N);
  k_count_deg<<<(E + T - 1) / T, T, 0, stream>>>(ei + (size_t)E, deg, E);
  k_rsqrt    <<<(N + T - 1) / T, T, 0, stream>>>(deg, N);

  // 2) hop 1: x -> h1
  k_selfloop    <<<((size_t)N * 32 + T - 1) / T, T, 0, stream>>>(x, deg, h1, N);
  k_edge_scatter<<<((size_t)E * 32 + T - 1) / T, T, 0, stream>>>(ei, deg, x, h1, E);
  // 3) hop 2: h1 -> h2
  k_selfloop    <<<((size_t)N * 32 + T - 1) / T, T, 0, stream>>>(h1, deg, h2, N);
  k_edge_scatter<<<((size_t)E * 32 + T - 1) / T, T, 0, stream>>>(ei, deg, h1, h2, E);

  // 4) pack operands into WMMA layouts
  int wpThreads = ntiles * ksteps * 512;
  k_pack_W<<<(wpThreads + T - 1) / T, T, 0, stream>>>(W, Wp, Cout, Cin);
  int apThreads = mtilesPad * ksteps * 512;
  k_pack_A<<<(apThreads + T - 1) / T, T, 0, stream>>>(h2, Ap, N, Cin, mtilesPad);

  // 5) GEMM + bias + LeakyReLU (overwrites all of d_out after packs consumed h)
  int mpairs = mtilesPad >> 1;
  int nquads = Cout >> 6;
  int waves  = mpairs * nquads;
  int blocks = (waves * 32 + T - 1) / T;
  k_gemm_wmma<<<blocks, T, 0, stream>>>(Ap, Wp, bias, out, N, Cout, Cin, mpairs);
}
